// TtMultiHeadAttentionModel_64158221467860
// MI455X (gfx1250) — matrix operand order repaired and checked by reading, not compiled, not run on hardware
//
#include <hip/hip_runtime.h>
#include <hip/hip_bf16.h>

#define B_  8
#define S_  512
#define H_  1024
#define NH_ 16
#define DH_ 64

typedef __attribute__((ext_vector_type(16))) __bf16        v16bf;
typedef __attribute__((ext_vector_type(8)))  float         v8f;
typedef __attribute__((ext_vector_type(8)))  unsigned short v8us;
typedef int v4i_ __attribute__((vector_size(4 * sizeof(int))));
typedef __attribute__((address_space(3))) v4i_* lds_v4i_ptr;

union V16 {
  v16bf bf;
  struct { v8us lo; v8us hi; } h;
};

// gfx1250 async global->LDS staging (ASYNCcnt-tracked, no VGPR round trip).
// Guarded: falls back to load+ds_store if the builtins aren't declared.
#if __has_builtin(__builtin_amdgcn_global_load_async_to_lds_b128) && \
    __has_builtin(__builtin_amdgcn_s_wait_asynccnt)
#define USE_ASYNC_LDS 1
#else
#define USE_ASYNC_LDS 0
#endif

__device__ inline void stage16(const unsigned short* g, unsigned short* l) {
#if USE_ASYNC_LDS
  // ROCm clang-22 signature (from diagnostics): (v4i* src, v4i[/AS3]* lds, Ii, Ii)
  __builtin_amdgcn_global_load_async_to_lds_b128((v4i_*)(g), (lds_v4i_ptr)(l),
                                                 0, 0);
#else
  *(v8us*)l = *(const v8us*)g;
#endif
}
__device__ inline void stage_fence() {
#if USE_ASYNC_LDS
  __builtin_amdgcn_s_wait_asynccnt(0);
#endif
}

// f32 -> bf16 bits, round-to-nearest-even
__device__ inline unsigned short f2bf(float f) {
  union { float f; unsigned int u; } x; x.f = f;
  unsigned int r = x.u + 0x7FFFu + ((x.u >> 16) & 1u);
  return (unsigned short)(r >> 16);
}

__device__ inline v8f wmma_bf16(v16bf a, v16bf b, v8f c) {
  // D(f32 16x16) = A(bf16 16x32) * B(bf16 32x16) + C
  return __builtin_amdgcn_wmma_f32_16x16x32_bf16(false, a, false, b, (short)0, c,
                                                 false, false);
}

// A-tile 16x32 bf16, row-major source with leading dim ld.
// ISA layout: lanes 0-15 hold row (lane&15), K {0..7,16..23};
//             lanes 16-31 hold same row,     K {8..15,24..31}.
__device__ inline v16bf load_a_tile(const unsigned short* base, int ld) {
  int lane = threadIdx.x & 31;
  int row = lane & 15, hi = lane >> 4;
  const unsigned short* p = base + (size_t)row * ld + hi * 8;
  V16 r;
  r.h.lo = *(const v8us*)(p);
  r.h.hi = *(const v8us*)(p + 16);
  return r.bf;
}

// B-tile 32x16 bf16 loaded from the TRANSPOSED operand Bt = B^T, [N][K] row-major.
// ISA layout: lanes 0-15: column (lane&15), K {0..7,16..23}; lanes 16-31: K {8..15,24..31}.
// Works on global or LDS pointers (addrspace inferred -> ds_load_b128 for LDS).
__device__ inline v16bf load_bt_tile(const unsigned short* base, int ld) {
  int lane = threadIdx.x & 31;
  int n = lane & 15, hi = lane >> 4;
  const unsigned short* p = base + (size_t)n * ld + hi * 8;
  V16 r;
  r.h.lo = *(const v8us*)(p);
  r.h.hi = *(const v8us*)(p + 16);
  return r.bf;
}

// ---------------------------------------------------------------------------
// Kernel 1: activation f32 [B*S, H] -> bf16 (same layout)
__global__ __launch_bounds__(256) void cvt_x(const float* __restrict__ in,
                                             unsigned short* __restrict__ out) {
  size_t i = ((size_t)blockIdx.x * 256 + threadIdx.x) * 4;
  float4 f = *(const float4*)(in + i);
  union { unsigned short s[4]; uint2 u; } o;
  o.s[0] = f2bf(f.x); o.s[1] = f2bf(f.y); o.s[2] = f2bf(f.z); o.s[3] = f2bf(f.w);
  *(uint2*)(out + i) = o.u;
}

// ---------------------------------------------------------------------------
// Kernel 2: Wt[m*H + o][i] = W_m[i][o] as bf16 (fused transposed weight [3H][H])
__global__ __launch_bounds__(256) void transpose_cvt(const float* __restrict__ wq,
                                                     const float* __restrict__ wk,
                                                     const float* __restrict__ wv,
                                                     unsigned short* __restrict__ Wt) {
  __shared__ float tile[32][33];
  int m = blockIdx.z;
  const float* src = (m == 0) ? wq : (m == 1) ? wk : wv;
  int o0 = blockIdx.x * 32, i0 = blockIdx.y * 32;
  int tx = threadIdx.x, ty = threadIdx.y;   // 32 x 8
#pragma unroll
  for (int r = 0; r < 32; r += 8)
    tile[ty + r][tx] = src[(size_t)(i0 + ty + r) * H_ + o0 + tx];
  __syncthreads();
#pragma unroll
  for (int r = 0; r < 32; r += 8)
    Wt[(size_t)(m * H_ + o0 + ty + r) * H_ + i0 + tx] = f2bf(tile[tx][ty + r]);
}

// ---------------------------------------------------------------------------
// Kernel 3: fused QKV GEMM. Block = 8 row-tiles (128 rows) x one 64-feature
// group. Weight chunk (64x32) staged in double-buffered LDS via async copies,
// shared by all 8 waves. A-tile software-pipelined in registers.
__global__ __launch_bounds__(256) void qkv_gemm(const unsigned short* __restrict__ Xb,   // [4096][1024]
                                                const unsigned short* __restrict__ Wt,   // [3072][1024]
                                                const float* __restrict__ bq,
                                                const float* __restrict__ bk,
                                                const float* __restrict__ bv,
                                                unsigned short* __restrict__ Qw,
                                                unsigned short* __restrict__ Kw,
                                                unsigned short* __restrict__ Vt) {
  __shared__ __align__(16) unsigned short Bsh[2][64][32];
  int tid = threadIdx.x;
  int wave = tid >> 5, lane = tid & 31;
  int n = lane & 15, hi = lane >> 4;
  int mgroup = blockIdx.x / 48;            // 32 groups of 8 row-tiles
  int n64 = blockIdx.x % 48;               // 48 feature groups of 64
  int mtile = mgroup * 8 + wave;

  const unsigned short* Abase = Xb + (size_t)mtile * 16 * H_;
  const unsigned short* Bbase = Wt + (size_t)(n64 * 64) * H_;
  int brow = tid >> 2, bpart = tid & 3;    // 64 rows x 4 x 16B = 4KB chunk

  // stage k-chunk 0
  stage16(Bbase + (size_t)brow * H_ + bpart * 8, &Bsh[0][brow][bpart * 8]);

  v8f acc[4];
#pragma unroll
  for (int c = 0; c < 4; c++) { v8f z = {}; acc[c] = z; }

  v16bf a = load_a_tile(Abase, H_);
  stage_fence();
  __syncthreads();

  for (int k = 0, it = 0; k < H_; k += 32, ++it) {
    int cur = it & 1;
    v16bf an = a;
    if (k + 32 < H_) {
      // stage next weight chunk into the other LDS buffer (async)
      stage16(Bbase + (size_t)brow * H_ + (k + 32) + bpart * 8,
              &Bsh[cur ^ 1][brow][bpart * 8]);
      // prefetch next A-tile into registers
      an = load_a_tile(Abase + k + 32, H_);
    }
#pragma unroll
    for (int c = 0; c < 4; c++)
      acc[c] = wmma_bf16(a, load_bt_tile(&Bsh[cur][c * 16][0], 32), acc[c]);
    a = an;
    stage_fence();
    __syncthreads();
  }

#pragma unroll
  for (int c = 0; c < 4; c++) {
    int f = n64 * 64 + c * 16 + n;            // 0..3071
    int m = f >> 10, oo = f & 1023;
    const float* bias = (m == 0) ? bq : (m == 1) ? bk : bv;
    float bval = bias[oo];
    int h = oo >> 6, d = oo & 63;
#pragma unroll
    for (int v = 0; v < 8; v++) {
      int r = mtile * 16 + v + 8 * hi;        // global row (C-layout: M=v / v+8)
      int b = r >> 9, s = r & 511;
      unsigned short val = f2bf(acc[c][v] + bval);
      if (m == 2)
        Vt[(((size_t)b * NH_ + h) * DH_ + d) * S_ + s] = val;
      else if (m == 1)
        Kw[(((size_t)b * NH_ + h) * S_ + s) * DH_ + d] = val;
      else
        Qw[(((size_t)b * NH_ + h) * S_ + s) * DH_ + d] = val;
    }
  }
}

// ---------------------------------------------------------------------------
// Kernel 4: flash attention. Block = 8 q-tiles of the SAME (b,h): K/V chunks
// async-staged into double-buffered LDS and shared by all 8 waves. Mask row
// staged once. Online softmax; P re-laid to A-format via per-wave LDS.
__global__ __launch_bounds__(256) void flash_attn(const unsigned short* __restrict__ Q,   // [B*NH,S,DH]
                                                  const unsigned short* __restrict__ K,   // [B*NH,S,DH]
                                                  const unsigned short* __restrict__ Vt,  // [B*NH,DH,S]
                                                  const float* __restrict__ mask,         // [B,S]
                                                  float* __restrict__ out) {              // [B,S,H]
  __shared__ __align__(16) unsigned short Ksh[2][32][64];   // K rows x d
  __shared__ __align__(16) unsigned short Vsh[2][64][32];   // d rows x k
  __shared__ __align__(16) unsigned short plds[8][16][40];  // per-wave P tile
  __shared__ __align__(16) float Msh[S_];                   // mask row
  int tid = threadIdx.x;
  int lane = tid & 31;
  int wave = tid >> 5;
  int n = lane & 15, hi = lane >> 4;
  int wid = blockIdx.x * 8 + wave;   // 0..4095
  int qt = wid & 31;                 // S/16 q-tiles; same bh across the block
  int bh = wid >> 5;                 // b*NH + h
  int b = bh >> 4, h = bh & 15;

  const unsigned short* Qb = Q + ((size_t)bh * S_ + qt * 16) * DH_;
  const unsigned short* Kb = K + (size_t)bh * S_ * DH_;
  const unsigned short* Vb = Vt + (size_t)bh * DH_ * S_;
  const float* mb = mask + (size_t)b * S_;

  int kr = tid >> 3, kp = tid & 7;   // K chunk: 32 rows x 8 x 16B
  int vr = tid >> 2, vp = tid & 3;   // V chunk: 64 rows x 4 x 16B

  // stage chunk 0 + mask row
  stage16(Kb + (size_t)kr * DH_ + kp * 8, &Ksh[0][kr][kp * 8]);
  stage16(Vb + (size_t)vr * S_ + vp * 8, &Vsh[0][vr][vp * 8]);
  Msh[tid] = mb[tid];
  Msh[tid + 256] = mb[tid + 256];

  v16bf qa0 = load_a_tile(Qb + 0, DH_);    // d 0..31
  v16bf qa1 = load_a_tile(Qb + 32, DH_);   // d 32..63

  v8f acc0 = {}, acc1 = {}, acc2 = {}, acc3 = {};
  float m_i[8], l_i[8];
#pragma unroll
  for (int v = 0; v < 8; v++) { m_i[v] = -1e30f; l_i[v] = 0.f; }

  stage_fence();
  __syncthreads();

  for (int kc = 0, it = 0; kc < S_; kc += 32, ++it) {
    int cur = it & 1;
    if (kc + 32 < S_) {
      stage16(Kb + (size_t)(kc + 32 + kr) * DH_ + kp * 8,
              &Ksh[cur ^ 1][kr][kp * 8]);
      stage16(Vb + (size_t)vr * S_ + (kc + 32) + vp * 8,
              &Vsh[cur ^ 1][vr][vp * 8]);
    }

    v8f s0 = {}, s1 = {};
    s0 = wmma_bf16(qa0, load_bt_tile(&Ksh[cur][0][0], DH_), s0);
    s0 = wmma_bf16(qa1, load_bt_tile(&Ksh[cur][0][32], DH_), s0);
    s1 = wmma_bf16(qa0, load_bt_tile(&Ksh[cur][16][0], DH_), s1);
    s1 = wmma_bf16(qa1, load_bt_tile(&Ksh[cur][16][32], DH_), s1);

    float mk0 = Msh[kc + n];
    float mk1 = Msh[kc + 16 + n];
    float tmax[8], p0[8], p1[8];
#pragma unroll
    for (int v = 0; v < 8; v++) {
      p0[v] = s0[v] * 0.125f + mk0;     // 1/sqrt(64)
      p1[v] = s1[v] * 0.125f + mk1;
      tmax[v] = fmaxf(p0[v], p1[v]);
    }
#pragma unroll
    for (int off = 1; off < 16; off <<= 1) {
#pragma unroll
      for (int v = 0; v < 8; v++)
        tmax[v] = fmaxf(tmax[v], __shfl_xor(tmax[v], off, 16));
    }

    float alpha[8], psum[8];
#pragma unroll
    for (int v = 0; v < 8; v++) {
      float mn = fmaxf(m_i[v], tmax[v]);
      alpha[v] = __expf(m_i[v] - mn);
      m_i[v] = mn;
      p0[v] = __expf(p0[v] - mn);
      p1[v] = __expf(p1[v] - mn);
      psum[v] = p0[v] + p1[v];
    }
#pragma unroll
    for (int off = 1; off < 16; off <<= 1) {
#pragma unroll
      for (int v = 0; v < 8; v++)
        psum[v] += __shfl_xor(psum[v], off, 16);
    }
#pragma unroll
    for (int v = 0; v < 8; v++) {
      l_i[v] = l_i[v] * alpha[v] + psum[v];
      acc0[v] *= alpha[v]; acc1[v] *= alpha[v];
      acc2[v] *= alpha[v]; acc3[v] *= alpha[v];
    }

    // C-layout -> A-layout transpose of P through LDS (per-wave; DS in-order)
#pragma unroll
    for (int v = 0; v < 8; v++) {
      plds[wave][v + 8 * hi][n] = f2bf(p0[v]);
      plds[wave][v + 8 * hi][16 + n] = f2bf(p1[v]);
    }
    V16 pa;
    {
      const unsigned short* pr = &plds[wave][n][0];   // A row = lane&15
      pa.h.lo = *(const v8us*)(pr + hi * 8);          // K {0..7} or {8..15}
      pa.h.hi = *(const v8us*)(pr + 16 + hi * 8);     // K {16..23} or {24..31}
    }

    acc0 = wmma_bf16(pa.bf, load_bt_tile(&Vsh[cur][0][0], 32), acc0);
    acc1 = wmma_bf16(pa.bf, load_bt_tile(&Vsh[cur][16][0], 32), acc1);
    acc2 = wmma_bf16(pa.bf, load_bt_tile(&Vsh[cur][32][0], 32), acc2);
    acc3 = wmma_bf16(pa.bf, load_bt_tile(&Vsh[cur][48][0], 32), acc3);

    stage_fence();
    __syncthreads();
  }

#pragma unroll
  for (int v = 0; v < 8; v++) {
    float invl = 1.0f / l_i[v];
    int srow = qt * 16 + v + 8 * hi;
    float* orow = out + ((size_t)b * S_ + srow) * H_ + h * DH_;
    orow[n]      = acc0[v] * invl;
    orow[16 + n] = acc1[v] * invl;
    orow[32 + n] = acc2[v] * invl;
    orow[48 + n] = acc3[v] * invl;
  }
}

// ---------------------------------------------------------------------------
extern "C" void kernel_launch(void* const* d_in, const int* in_sizes, int n_in,
                              void* d_out, int out_size, void* d_ws, size_t ws_size,
                              hipStream_t stream) {
  (void)in_sizes; (void)n_in; (void)out_size; (void)ws_size;
  const float* act = (const float*)d_in[0];
  const float* amask = (const float*)d_in[1];
  const float* wq = (const float*)d_in[2];
  const float* bq = (const float*)d_in[3];
  const float* wk = (const float*)d_in[4];
  const float* bk = (const float*)d_in[5];
  const float* wv = (const float*)d_in[6];
  const float* bv = (const float*)d_in[7];
  float* out = (float*)d_out;

  // Workspace layout (bf16 halves): 8 + 6 + 8 + 8 + 8 = 38 MB
  unsigned short* Xb = (unsigned short*)d_ws;                   // [4096][1024]
  unsigned short* Wt = Xb + (size_t)B_ * S_ * H_;               // [3072][1024]
  unsigned short* Qw = Wt + (size_t)3 * H_ * H_;                // [B,NH,S,DH]
  unsigned short* Kw = Qw + (size_t)B_ * NH_ * S_ * DH_;        // [B,NH,S,DH]
  unsigned short* Vt = Kw + (size_t)B_ * NH_ * S_ * DH_;        // [B,NH,DH,S]

  cvt_x<<<(B_ * S_ * H_) / (256 * 4), 256, 0, stream>>>(act, Xb);
  transpose_cvt<<<dim3(H_ / 32, H_ / 32, 3), dim3(32, 8), 0, stream>>>(wq, wk, wv, Wt);
  qkv_gemm<<<32 * 48, 256, 0, stream>>>(Xb, Wt, bq, bk, bv, Qw, Kw, Vt);
  flash_attn<<<(B_ * NH_ * (S_ / 16)) / 8, 256, 0, stream>>>(Qw, Kw, Vt, amask, out);
}
